// AugmentedNeuralODE_24687472017936
// MI455X (gfx1250) — matrix operand order repaired and checked by reading, not compiled, not run on hardware
//
#include <hip/hip_runtime.h>
#include <cstddef>
#include <cstdint>

// ---------------------------------------------------------------------------
// CDNA5 / gfx1250 WMMA building blocks (wave32, V_WMMA_F32_16X16X32_F16)
// ---------------------------------------------------------------------------
typedef __attribute__((ext_vector_type(16))) _Float16 v16h;
typedef __attribute__((ext_vector_type(8)))  float    v8f;
typedef __attribute__((ext_vector_type(4)))  unsigned v4u;

union Frag16 { v16h v; v4u q[2]; };

// Opaque zero: prevents LICM from hoisting weight-fragment loads across the
// sequential (time-step / RK-stage) loops, which previously caused VGPR
// exhaustion and scratch spill/reload inside the hot WMMA loop. Weights are
// WGP$/L2 resident, so re-fetching beats scratch round-trips.
__device__ __forceinline__ int opaque_zero() {
  int z = 0;
  asm volatile("" : "+v"(z));
  return z;
}

// A-matrix 16x32 f16 fragment from an LDS row-major [16][ldk] f16 tile.
// ISA layout: lanes 0-15 -> M=lane, lanes 16-31 -> M=lane-16.
// half j: k = 16*half + j
// => dwords: u[0..3] = row32[8h .. 8h+3], u[4..7] = row32[8h+4 .. 8h+7]
__device__ __forceinline__ v16h ld_afrag(const _Float16* tile, int ldk, int kt) {
  const int lane = threadIdx.x & 31;
  const int m = lane & 15, h = lane >> 4;
  const unsigned* r32 = (const unsigned*)(tile + m * ldk + kt * 32);
  Frag16 U;
  U.q[0] = *(const v4u*)(r32 + 8 * h);
  U.q[1] = *(const v4u*)(r32 + 8 * h + 4);
  return U.v;
}

// B-matrix 32x16 f16 fragment from pre-packed weights:
// element j of lane: k = kt*32 + 16*(lane>>4) + j, n = nt*16 + (lane&15)
// packed so each lane reads 16 contiguous halves (2x b128).
__device__ __forceinline__ v16h ld_bfrag(const _Float16* wf, int nt, int kt, int ktc) {
  const int lane = threadIdx.x & 31;
  const _Float16* p = wf + ((size_t)(((nt * ktc + kt) << 5) + lane) << 4);
  const v4u* q = (const v4u*)p;
  Frag16 U;
  U.q[0] = q[0];
  U.q[1] = q[1];
  return U.v;
}

__device__ __forceinline__ v8f wmma_f16(v16h a, v16h b, v8f c) {
  return __builtin_amdgcn_wmma_f32_16x16x32_f16(false, a, false, b, (short)0, c,
                                                false, false);
}

__device__ __forceinline__ float sigmoidf_(float x) {
  return 1.0f / (1.0f + __expf(-x));
}

// ---------------------------------------------------------------------------
// Tsit5 tableau
// ---------------------------------------------------------------------------
__constant__ float TSIT_A[6][5] = {
    {0.f, 0.f, 0.f, 0.f, 0.f},
    {0.161f, 0.f, 0.f, 0.f, 0.f},
    {-0.008480655492356989f, 0.335480655492357f, 0.f, 0.f, 0.f},
    {2.8971530571054935f, -6.359448489975075f, 4.3622954328695815f, 0.f, 0.f},
    {5.325864828439257f, -11.748883564062828f, 7.4955393428898365f,
     -0.09249506636175525f, 0.f},
    {5.86145544294642f, -12.92096931784711f, 8.159367898576159f,
     -0.071584973281401f, -0.028269050394068383f}};
__constant__ float TSIT_B[6] = {0.09646076681806523f, 0.01f,
                                0.4798896504144996f,  1.379008574103742f,
                                -3.290069515436081f,  2.324710524099774f};

// ---------------------------------------------------------------------------
// Weight packing: f32 row-major W[N][K]  ->  f16 B-fragment tiles
// ---------------------------------------------------------------------------
__global__ void k_pack(const float* __restrict__ W, _Float16* __restrict__ dst,
                       int N, int K) {
  const int ktc = K >> 5;
  const int nt = blockIdx.x / ktc;
  const int kt = blockIdx.x - nt * ktc;
  const int lane = threadIdx.x;
  const int n = nt * 16 + (lane & 15);
  const int kb = kt * 32 + ((lane >> 4) << 4);
  _Float16* o = dst + ((size_t)(((nt * ktc + kt) << 5) + lane) << 4);
  const float* src = W + (size_t)n * K + kb;
#pragma unroll
  for (int j = 0; j < 16; ++j) o[j] = (_Float16)src[j];
}

// ---------------------------------------------------------------------------
// GRU recurrence: persistent workgroup per 16 batch rows, 64 sequential steps.
// Gates computed with WMMA (x@Wih^T and h@Whh^T, n-gate parts kept separate).
// wih fragments (12 frags/wave) are allowed to live in registers across the
// whole sequence; whh fragments (48 frags/wave) are re-streamed from cache
// each step via the laundered pointer to avoid scratch spills.
// ---------------------------------------------------------------------------
__global__ __launch_bounds__(256) void k_gru(
    const float* __restrict__ yi, const _Float16* __restrict__ wih,
    const _Float16* __restrict__ whh, const float* __restrict__ gb,
    const float* __restrict__ gbn, _Float16* __restrict__ h_final) {
  __shared__ _Float16 xbuf[16 * 64];
  __shared__ _Float16 hbuf[2][16 * 256];
  __shared__ float hf32[16 * 256];

  const int tid = threadIdx.x;
  const int b0 = blockIdx.x * 16;
  for (int i = tid; i < 16 * 256; i += 256) {
    hbuf[0][i] = (_Float16)0.f;
    hf32[i] = 0.f;
  }
  __syncthreads();

  const int wid = tid >> 5, lane = tid & 31;
  const int nloc = lane & 15, hh = lane >> 4;

  for (int s = 0; s < 64; ++s) {
    // xs[s] = yi[:, 63-s, :]  (reversed sequence), converted to f16 in LDS
    for (int i = tid; i < 16 * 64; i += 256) {
      int rr = i >> 6, cc = i & 63;
      xbuf[i] = (_Float16)yi[((size_t)(b0 + rr) * 64 + (63 - s)) * 64 + cc];
    }
    __syncthreads();
    const _Float16* hc = hbuf[s & 1];
    _Float16* hn = hbuf[(s + 1) & 1];
    const _Float16* whhp = whh + opaque_zero();  // fresh per step: no hoist

#pragma unroll
    for (int tt = 0; tt < 2; ++tt) {
      const int t = wid * 2 + tt;  // hidden-unit n-tile [0,16)
      v8f aR = {}, aZ = {}, aNi = {}, aNh = {};
#pragma unroll
      for (int kt = 0; kt < 2; ++kt) {  // K = DATA = 64
        v16h ax = ld_afrag(xbuf, 64, kt);
        aR = wmma_f16(ax, ld_bfrag(wih, t, kt, 2), aR);
        aZ = wmma_f16(ax, ld_bfrag(wih, 16 + t, kt, 2), aZ);
        aNi = wmma_f16(ax, ld_bfrag(wih, 32 + t, kt, 2), aNi);
      }
#pragma unroll
      for (int kt = 0; kt < 8; ++kt) {  // K = HIDDEN = 256
        v16h ah = ld_afrag(hc, 256, kt);
        aR = wmma_f16(ah, ld_bfrag(whhp, t, kt, 8), aR);
        aZ = wmma_f16(ah, ld_bfrag(whhp, 16 + t, kt, 8), aZ);
        aNh = wmma_f16(ah, ld_bfrag(whhp, 32 + t, kt, 8), aNh);
      }
      const int unit = t * 16 + nloc;
      const float br = gb[unit], bz = gb[256 + unit];
      const float bni = gb[512 + unit], bnn = gbn[unit];
#pragma unroll
      for (int r = 0; r < 8; ++r) {
        const int m = r + 8 * hh;
        float rg = sigmoidf_(aR[r] + br);
        float zg = sigmoidf_(aZ[r] + bz);
        float ng = tanhf(aNi[r] + bni + rg * (aNh[r] + bnn));
        float ho = hf32[m * 256 + unit];
        float hv = ng + zg * (ho - ng);
        hf32[m * 256 + unit] = hv;
        hn[m * 256 + unit] = (_Float16)hv;
      }
    }
    __syncthreads();
  }
  for (int i = tid; i < 16 * 256; i += 256)
    h_final[(size_t)(b0 + (i >> 8)) * 256 + (i & 255)] = (_Float16)hf32[i];
}

// ---------------------------------------------------------------------------
// Generic tiled WMMA MLP layer: out = act(in @ W^T + b)
// Block owns 16 rows; 8 waves split the N/16 column tiles.
// ---------------------------------------------------------------------------
template <int K, int N, bool TANH, bool O16, bool O32, bool TB>
__global__ __launch_bounds__(256) void k_mlp(const _Float16* __restrict__ inA,
                                             const _Float16* __restrict__ wf,
                                             const float* __restrict__ bias,
                                             _Float16* __restrict__ out16,
                                             float* __restrict__ out32) {
  __shared__ _Float16 atile[16 * K];
  const int tid = threadIdx.x;
  {
    const unsigned* s = (const unsigned*)(inA + (size_t)blockIdx.x * 16 * K);
    unsigned* d = (unsigned*)atile;
    constexpr int cnt = (16 * K) / 2;
    for (int i = tid; i < cnt; i += 256) d[i] = s[i];
  }
  __syncthreads();
  const int wid = tid >> 5, lane = tid & 31;
  const int nloc = lane & 15, hh = lane >> 4;
  constexpr int ktc = K / 32, ntiles = N / 16;
  for (int nt = wid; nt < ntiles; nt += 8) {
    v8f acc = {};
#pragma unroll
    for (int kt = 0; kt < ktc; ++kt)
      acc = wmma_f16(ld_afrag(atile, K, kt), ld_bfrag(wf, nt, kt, ktc), acc);
    const int n = nt * 16 + nloc;
    const float bb = bias[n];
#pragma unroll
    for (int r = 0; r < 8; ++r) {
      const int row = blockIdx.x * 16 + r + 8 * hh;
      float v = acc[r] + bb;
      if (TANH) v = tanhf(v);
      if (O16) out16[(size_t)row * N + n] = (_Float16)v;
      if (O32) {
        if (TB) {  // rows ordered [t][b]; emit [b][t][n]
          const int t = row >> 9, b = row & 511;
          out32[((size_t)b * 32 + t) * N + n] = v;
        } else {
          out32[(size_t)row * N + n] = v;
        }
      }
    }
  }
}

// ---------------------------------------------------------------------------
// In-workgroup MLP layers for the ODE integrator (LDS -> LDS)
// ---------------------------------------------------------------------------
template <int K, int N, bool TANH>
__device__ __forceinline__ void wg_layer16(const _Float16* in, _Float16* out,
                                           const _Float16* wf,
                                           const float* bias) {
  const int wid = threadIdx.x >> 5, lane = threadIdx.x & 31;
  const int nloc = lane & 15, hh = lane >> 4;
  constexpr int ktc = K / 32, ntiles = N / 16;
  for (int nt = wid; nt < ntiles; nt += 8) {
    v8f acc = {};
#pragma unroll
    for (int kt = 0; kt < ktc; ++kt)
      acc = wmma_f16(ld_afrag(in, K, kt), ld_bfrag(wf, nt, kt, ktc), acc);
    const int n = nt * 16 + nloc;
    const float bb = bias[n];
#pragma unroll
    for (int r = 0; r < 8; ++r) {
      float v = acc[r] + bb;
      if (TANH) v = tanhf(v);
      out[(r + 8 * hh) * N + n] = (_Float16)v;
    }
  }
}

template <int K, int N>
__device__ __forceinline__ void wg_layer32(const _Float16* in, float* out,
                                           const _Float16* wf,
                                           const float* bias) {
  const int wid = threadIdx.x >> 5, lane = threadIdx.x & 31;
  const int nloc = lane & 15, hh = lane >> 4;
  constexpr int ktc = K / 32, ntiles = N / 16;
  for (int nt = wid; nt < ntiles; nt += 8) {
    v8f acc = {};
#pragma unroll
    for (int kt = 0; kt < ktc; ++kt)
      acc = wmma_f16(ld_afrag(in, K, kt), ld_bfrag(wf, nt, kt, ktc), acc);
    const int n = nt * 16 + nloc;
    const float bb = bias[n];
#pragma unroll
    for (int r = 0; r < 8; ++r) out[(r + 8 * hh) * N + n] = acc[r] + bb;
  }
}

// ---------------------------------------------------------------------------
// Tsit5 integrator: persistent workgroup per 16 batch rows. y state (f32) and
// all six stage derivatives live in LDS across all 31 intervals x 2 substeps.
// Weight pointers are laundered per stage so fragments are re-streamed from
// WGP$/L2 instead of being hoisted into (and spilled out of) registers.
// ---------------------------------------------------------------------------
__global__ __launch_bounds__(256) void k_ode(
    const float* __restrict__ ts, const float* __restrict__ y0,
    const _Float16* __restrict__ wf0, const _Float16* __restrict__ wf1,
    const _Float16* __restrict__ wf2, const _Float16* __restrict__ wf3,
    const float* __restrict__ fb0, const float* __restrict__ fb1,
    const float* __restrict__ fb2, const float* __restrict__ fb3,
    _Float16* __restrict__ ys) {
  __shared__ float y32[16 * 128];
  __shared__ _Float16 yst[16 * 128];
  __shared__ _Float16 actP[16 * 256];
  __shared__ _Float16 actQ[16 * 256];
  __shared__ float kb[6][16 * 128];

  const int tid = threadIdx.x;
  const int b0 = blockIdx.x * 16;
  for (int i = tid; i < 2048; i += 256) y32[i] = y0[(size_t)b0 * 128 + i];
  __syncthreads();

  for (int t = 0; t < 31; ++t) {
    const float dt = (ts[t + 1] - ts[t]) * 0.5f;  // / SUBSTEPS
    for (int sub = 0; sub < 2; ++sub) {
      for (int stg = 0; stg < 6; ++stg) {
        for (int i = tid; i < 2048; i += 256) {
          float v = y32[i];
          for (int j = 0; j < stg; ++j) v += dt * TSIT_A[stg][j] * kb[j][i];
          yst[i] = (_Float16)v;
        }
        __syncthreads();
        const int z = opaque_zero();  // fresh per stage: no cross-stage hoist
        const _Float16* w0p = wf0 + z;
        const _Float16* w1p = wf1 + z;
        const _Float16* w2p = wf2 + z;
        const _Float16* w3p = wf3 + z;
        wg_layer16<128, 256, true>(yst, actP, w0p, fb0);
        __syncthreads();
        wg_layer16<256, 256, true>(actP, actQ, w1p, fb1);
        __syncthreads();
        wg_layer16<256, 256, true>(actQ, actP, w2p, fb2);
        __syncthreads();
        wg_layer32<256, 128>(actP, kb[stg], w3p, fb3);
        __syncthreads();
      }
      for (int i = tid; i < 2048; i += 256) {
        float a = 0.f;
#pragma unroll
        for (int j = 0; j < 6; ++j) a += TSIT_B[j] * kb[j][i];
        y32[i] += dt * a;
      }
      __syncthreads();
    }
    _Float16* dst = ys + ((size_t)(t + 1) * 512 + b0) * 128;
    for (int i = tid; i < 2048; i += 256) dst[i] = (_Float16)y32[i];
  }
}

// ---------------------------------------------------------------------------
// Host orchestration
// ---------------------------------------------------------------------------
extern "C" void kernel_launch(void* const* d_in, const int* in_sizes, int n_in,
                              void* d_out, int out_size, void* d_ws,
                              size_t ws_size, hipStream_t stream) {
  (void)in_sizes; (void)n_in; (void)out_size; (void)ws_size;
  const float* ts = (const float*)d_in[0];
  const float* yi = (const float*)d_in[1];
  const float* gwih = (const float*)d_in[2];
  const float* gwhh = (const float*)d_in[3];
  const float* gb = (const float*)d_in[4];
  const float* gbn = (const float*)d_in[5];
  const float* fW0 = (const float*)d_in[6];  const float* fb0 = (const float*)d_in[7];
  const float* fW1 = (const float*)d_in[8];  const float* fb1 = (const float*)d_in[9];
  const float* fW2 = (const float*)d_in[10]; const float* fb2 = (const float*)d_in[11];
  const float* fW3 = (const float*)d_in[12]; const float* fb3 = (const float*)d_in[13];
  const float* hW0 = (const float*)d_in[14]; const float* hb0 = (const float*)d_in[15];
  const float* hW1 = (const float*)d_in[16]; const float* hb1 = (const float*)d_in[17];
  const float* hW2 = (const float*)d_in[18]; const float* hb2 = (const float*)d_in[19];
  const float* oW0 = (const float*)d_in[20]; const float* ob0 = (const float*)d_in[21];
  const float* oW1 = (const float*)d_in[22]; const float* ob1 = (const float*)d_in[23];
  const float* oW2 = (const float*)d_in[24]; const float* ob2 = (const float*)d_in[25];

  char* ws = (char*)d_ws;
  size_t off = 0;
  auto a16 = [&](size_t halves) { _Float16* p = (_Float16*)(ws + off); off += halves * 2; return p; };
  auto a32 = [&](size_t n)      { float* p = (float*)(ws + off); off += n * 4; return p; };

  _Float16* wih_f = a16(768 * 64);
  _Float16* whh_f = a16(768 * 256);
  _Float16* f0_f = a16(256 * 128);
  _Float16* f1_f = a16(256 * 256);
  _Float16* f2_f = a16(256 * 256);
  _Float16* f3_f = a16(128 * 256);
  _Float16* h0_f = a16(256 * 256);
  _Float16* h1_f = a16(256 * 256);
  _Float16* h2_f = a16(128 * 256);
  _Float16* o0_f = a16(256 * 128);
  _Float16* o1_f = a16(256 * 256);
  _Float16* o2_f = a16(64 * 256);
  _Float16* hfin = a16(512 * 256);
  float* y0f = a32(512 * 128);
  _Float16* ysb = a16((size_t)32 * 512 * 128);
  _Float16* tA = a16((size_t)16384 * 256);
  _Float16* tB = a16((size_t)16384 * 256);

  auto pk = [&](const float* W, _Float16* dst, int N, int K) {
    k_pack<<<dim3((N / 16) * (K / 32)), dim3(32), 0, stream>>>(W, dst, N, K);
  };
  pk(gwih, wih_f, 768, 64);
  pk(gwhh, whh_f, 768, 256);
  pk(fW0, f0_f, 256, 128);
  pk(fW1, f1_f, 256, 256);
  pk(fW2, f2_f, 256, 256);
  pk(fW3, f3_f, 128, 256);
  pk(hW0, h0_f, 256, 256);
  pk(hW1, h1_f, 256, 256);
  pk(hW2, h2_f, 128, 256);
  pk(oW0, o0_f, 256, 128);
  pk(oW1, o1_f, 256, 256);
  pk(oW2, o2_f, 64, 256);

  // GRU encoder (persistent: 32 WGs x 16 batch rows, 64 sequential steps)
  k_gru<<<dim3(32), dim3(256), 0, stream>>>(yi, wih_f, whh_f, gb, gbn, hfin);

  // h2o MLP: 256 -> 256 -> 256 -> 128 (tanh, tanh, identity); last layer also
  // seeds ys[t=0] (f16) and y0 (f32) for the integrator.
  k_mlp<256, 256, true, true, false, false><<<dim3(32), dim3(256), 0, stream>>>(hfin, h0_f, hb0, tA, nullptr);
  k_mlp<256, 256, true, true, false, false><<<dim3(32), dim3(256), 0, stream>>>(tA, h1_f, hb1, tB, nullptr);
  k_mlp<256, 128, false, true, true, false><<<dim3(32), dim3(256), 0, stream>>>(tB, h2_f, hb2, ysb, y0f);

  // Tsit5 integration (persistent workgroups; 372 WMMA MLP evals in-kernel)
  k_ode<<<dim3(32), dim3(256), 0, stream>>>(ts, y0f, f0_f, f1_f, f2_f, f3_f,
                                            fb0, fb1, fb2, fb3, ysb);

  // o2d linear MLP over all T*B rows: 128 -> 256 -> 256 -> 64, final layer
  // writes f32 directly to d_out in [B, T, DATA] order.
  k_mlp<128, 256, false, true, false, false><<<dim3(1024), dim3(256), 0, stream>>>(ysb, o0_f, ob0, tA, nullptr);
  k_mlp<256, 256, false, true, false, false><<<dim3(1024), dim3(256), 0, stream>>>(tA, o1_f, ob1, tB, nullptr);
  k_mlp<256, 64, false, false, true, true><<<dim3(1024), dim3(256), 0, stream>>>(tB, o2_f, ob2, nullptr, (float*)d_out);
}
